// HSTGNN_43430709297850
// MI455X (gfx1250) — compile-verified, matrix operand order checked by reading, not run on hardware
//
#include <hip/hip_runtime.h>
#include <hip/hip_bf16.h>
#include <stdint.h>

// Problem dims (fixed by the reference)
#define Bd 4
#define Cd 256
#define Nd 1024
#define Ld 12
#define Rd (Bd*Ld*Nd)      // 49152 rows in canonical [b,l,n,c] layout
#define BLd (Bd*Ld)        // 48 (b,l) slices

typedef __attribute__((ext_vector_type(16))) _Float16 v16h;
typedef __attribute__((ext_vector_type(8)))  _Float16 v8h;
typedef __attribute__((ext_vector_type(8)))  float    v8f;
typedef _Float16 half_t;

union V16U { v16h v; v8h h[2]; };

// gfx1250 async global->LDS copy (16B per lane), tracked by ASYNCcnt.
__device__ __forceinline__ void async_b128(unsigned lds_addr, const half_t* gptr)
{
    asm volatile("global_load_async_to_lds_b128 %0, %1, off"
                 :: "v"(lds_addr), "v"((unsigned long long)(uintptr_t)gptr)
                 : "memory");
}
__device__ __forceinline__ void wait_async0()
{
    asm volatile("s_wait_asynccnt 0x0" ::: "memory");
}

// ---------------------------------------------------------------------------
// WMMA GEMM:  C[M x Nn] (f32) = A[M x K] (f16, row-major) * B (f16, given
// COLUMN-major as BT[Nn x K]).  Block tile 64x64, K-step 32, 8 waves; each
// wave computes 16x32 of C (2 WMMA tiles).  Tiles are fetched with
// GLOBAL_LOAD_ASYNC_TO_LDS_B128 and pipelined against the WMMAs (ASYNCcnt).
// Batched via blockIdx.z with element strides sA/sB/sC (sA==0 shares A).
// All dims are multiples of 64/32 -> no bounds checks, EXEC stays all-1s.
// ---------------------------------------------------------------------------
__global__ __launch_bounds__(256)
void gemm_wmma(const half_t* __restrict__ A, const half_t* __restrict__ BT,
               float* __restrict__ C,
               int M, int Nn, int K, int lda, int ldbt, int ldc,
               long long sA, long long sB, long long sC)
{
    __shared__ __align__(16) half_t As[64][32];   // [m][k]
    __shared__ __align__(16) half_t Bs[64][32];   // [n][k]

    const int z = blockIdx.z;
    A  += (size_t)z * (size_t)sA;
    BT += (size_t)z * (size_t)sB;
    C  += (size_t)z * (size_t)sC;

    const int bm   = blockIdx.y * 64;
    const int bn   = blockIdx.x * 64;
    const int t    = threadIdx.x;
    const int wid  = t >> 5;          // 0..7
    const int lane = t & 31;
    const int wm   = wid >> 1;        // 0..3 (M direction)
    const int wn   = wid & 1;         // 0..1 (N direction)
    const int l15  = lane & 15;
    const int hi   = lane >> 4;       // half-wave select (K packing)

    v8f c0 = {};
    v8f c1 = {};

    // per-thread 16B copy slots: row = t>>2 (0..63), kcol = (t&3)*8
    const int row  = t >> 2;
    const int kcol = (t & 3) * 8;
    const half_t* agp = A  + (size_t)(bm + row) * (size_t)lda  + kcol;
    const half_t* bgp = BT + (size_t)(bn + row) * (size_t)ldbt + kcol;
    const unsigned aLds = (unsigned)(uintptr_t)&As[row][kcol];
    const unsigned bLds = (unsigned)(uintptr_t)&Bs[row][kcol];

    // prologue: tile 0 in flight
    async_b128(aLds, agp);
    async_b128(bLds, bgp);

    for (int k0 = 0; k0 < K; k0 += 32) {
        wait_async0();            // this wave's copies landed
        __syncthreads();          // everyone's copies landed

        // Fragment packing per CDNA5 ISA 16-bit A layout:
        //   element i: k = (i>>3)*16 + hi*8 + (i&7)  -> two contiguous v8h loads
        V16U af, bf0, bf1;
        const int mrow = wm * 16 + l15;
        af.h[0] = *(const v8h*)&As[mrow][hi * 8];
        af.h[1] = *(const v8h*)&As[mrow][16 + hi * 8];
        const int nc0 = wn * 32 + l15;
        bf0.h[0] = *(const v8h*)&Bs[nc0][hi * 8];
        bf0.h[1] = *(const v8h*)&Bs[nc0][16 + hi * 8];
        const int nc1 = nc0 + 16;
        bf1.h[0] = *(const v8h*)&Bs[nc1][hi * 8];
        bf1.h[1] = *(const v8h*)&Bs[nc1][16 + hi * 8];

        __syncthreads();          // all waves done reading LDS (dscnt waited)

        if (k0 + 32 < K) {        // overlap next tile's DMA with the WMMAs
            async_b128(aLds, agp + (k0 + 32));
            async_b128(bLds, bgp + (k0 + 32));
        }

        c0 = __builtin_amdgcn_wmma_f32_16x16x32_f16(false, af.v, false, bf0.v,
                                                    (short)0, c0, false, false);
        c1 = __builtin_amdgcn_wmma_f32_16x16x32_f16(false, af.v, false, bf1.v,
                                                    (short)0, c1, false, false);
    }

    // C/D layout: VGPR j -> row (j + 8*hi), col = lane&15
    const int crow0 = bm + wm * 16 + hi * 8;
    const int ccol0 = bn + wn * 32 + l15;
    #pragma unroll
    for (int j = 0; j < 8; ++j) {
        C[(size_t)(crow0 + j) * (size_t)ldc + ccol0]      = c0[j];
        C[(size_t)(crow0 + j) * (size_t)ldc + ccol0 + 16] = c1[j];
    }
}

// ---------------------------------------------------------------------------
// Pack x_in [b,c,n,l] -> Xf32 [b,l,n,c], Hcat f16 (row stride 512, cols 0..255),
// and XT16 [bl][c][n] (column-major B operand for the x1 GEMM).
// Coalesced on one side; scattered side soaked up by the 192MB L2.
// ---------------------------------------------------------------------------
__global__ void k_pack(const float* __restrict__ xin, float* __restrict__ xf32,
                       half_t* __restrict__ hcat, half_t* __restrict__ xt16, size_t n)
{
    size_t idx = (size_t)blockIdx.x * 256 + threadIdx.x;
    if (idx >= n) return;
    int c = (int)(idx & 255);
    size_t r = idx >> 8;
    int nn = (int)(r & (Nd - 1));
    int bl = (int)(r >> 10);
    int l = bl % Ld;
    int b = bl / Ld;
    float v = xin[((size_t)(b * Cd + c) * Nd + nn) * Ld + l];
    half_t h = (half_t)v;
    xf32[idx] = v;
    hcat[r * 512 + c] = h;
    xt16[(((size_t)bl * Cd + c) << 10) + nn] = h;
}

// Final unpack: y [b,l,n,c] -> out [b,c,n,l]
__global__ void k_unpack(const float* __restrict__ y, float* __restrict__ out, size_t n)
{
    size_t idx = (size_t)blockIdx.x * 256 + threadIdx.x;
    if (idx >= n) return;
    int l = (int)(idx % Ld);
    size_t tmp = idx / Ld;
    int nn = (int)(tmp & (Nd - 1));
    tmp >>= 10;
    int c = (int)(tmp & (Cd - 1));
    int b = (int)(tmp >> 8);
    out[idx] = y[((size_t)(b * Ld + l) * Nd + nn) * Cd + c];
}

// f32 -> f16 flat
__global__ void k_cvt(const float* __restrict__ s, half_t* __restrict__ d, size_t n)
{
    size_t i = (size_t)blockIdx.x * 256 + threadIdx.x;
    if (i < n) d[i] = (half_t)s[i];
}

// f32 -> f16 with transpose: src [srcRows x srcCols] row-major -> dst [srcCols x srcRows]
__global__ void k_cvtT(const float* __restrict__ s, half_t* __restrict__ d,
                       int srcRows, int srcCols, size_t n)
{
    size_t i = (size_t)blockIdx.x * 256 + threadIdx.x;
    if (i >= n) return;
    int r = (int)(i % srcRows);
    int c = (int)(i / srcRows);
    d[i] = (half_t)s[(size_t)r * srcCols + c];
}

// Wh f32 [bl][n][c] -> WhT f16 [bl][c][n]  (column-major B operand per slice)
__global__ void k_whT(const float* __restrict__ wh, half_t* __restrict__ whT, size_t n)
{
    size_t i = (size_t)blockIdx.x * 256 + threadIdx.x;
    if (i >= n) return;
    int nn = (int)(i & (Nd - 1));
    int c  = (int)((i >> 10) & (Cd - 1));
    int bl = (int)(i >> 18);
    whT[i] = (half_t)wh[(((size_t)bl << 10) + nn) * Cd + c];
}

// x1 f32 [R x 256] -> second half of Hcat (row stride 512, col offset 256)
__global__ void k_cvt_x1(const float* __restrict__ s, half_t* __restrict__ hcat, size_t n)
{
    size_t i = (size_t)blockIdx.x * 256 + threadIdx.x;
    if (i >= n) return;
    hcat[(i >> 8) * 512 + 256 + (i & 255)] = (half_t)s[i];
}

// elu(x) -> f16
__global__ void k_elu16(const float* __restrict__ s, half_t* __restrict__ d, size_t n)
{
    size_t i = (size_t)blockIdx.x * 256 + threadIdx.x;
    if (i >= n) return;
    float x = s[i];
    d[i] = (half_t)(x > 0.f ? x : (__expf(x) - 1.f));
}

// adp = softmax(relu(nv1 @ nv2), axis=1); one block per row, stored as f16
__global__ __launch_bounds__(256) void k_adp(const float* __restrict__ nv1,
                                             const float* __restrict__ nv2,
                                             half_t* __restrict__ adp)
{
    __shared__ float red[256];
    const int i = blockIdx.x;
    float a0 = nv1[i*6+0], a1 = nv1[i*6+1], a2 = nv1[i*6+2];
    float a3 = nv1[i*6+3], a4 = nv1[i*6+4], a5 = nv1[i*6+5];
    const int t = threadIdx.x;
    float ev[4];
    float mx = -3.4e38f;
    #pragma unroll
    for (int q = 0; q < 4; ++q) {
        int j = t + q * 256;
        float s = a0*nv2[j] + a1*nv2[Nd+j] + a2*nv2[2*Nd+j]
                + a3*nv2[3*Nd+j] + a4*nv2[4*Nd+j] + a5*nv2[5*Nd+j];
        s = fmaxf(s, 0.f);
        ev[q] = s;
        mx = fmaxf(mx, s);
    }
    red[t] = mx; __syncthreads();
    for (int st = 128; st > 0; st >>= 1) { if (t < st) red[t] = fmaxf(red[t], red[t+st]); __syncthreads(); }
    mx = red[0]; __syncthreads();
    float sum = 0.f;
    #pragma unroll
    for (int q = 0; q < 4; ++q) { ev[q] = __expf(ev[q] - mx); sum += ev[q]; }
    red[t] = sum; __syncthreads();
    for (int st = 128; st > 0; st >>= 1) { if (t < st) red[t] += red[t+st]; __syncthreads(); }
    const float inv = 1.f / red[0];
    half_t* dst = adp + (size_t)i * Nd;
    #pragma unroll
    for (int q = 0; q < 4; ++q) dst[t + q * 256] = (half_t)(ev[q] * inv);
}

// Wh1 = Wh @ a[:256], Wh2 = Wh @ a[256:]; one wave32 per row
__global__ __launch_bounds__(256) void k_dots(const float* __restrict__ wh,
                                              const float* __restrict__ a,
                                              float* __restrict__ o1, float* __restrict__ o2)
{
    const int wid = threadIdx.x >> 5, lane = threadIdx.x & 31;
    const int r = blockIdx.x * 8 + wid;
    const float* p = wh + (size_t)r * Cd;
    float s1 = 0.f, s2 = 0.f;
    #pragma unroll
    for (int q = 0; q < 8; ++q) {
        int tc = lane + q * 32;
        float w = p[tc];
        s1 += w * a[tc];
        s2 += w * a[Cd + tc];
    }
    #pragma unroll
    for (int m = 16; m > 0; m >>= 1) {
        s1 += __shfl_xor(s1, m, 32);
        s2 += __shfl_xor(s2, m, 32);
    }
    if (lane == 0) { o1[r] = s1; o2[r] = s2; }
}

// att[bl,i,:] = softmax_j( mask ? leaky(Wh1[i]+Wh2[j]) : -9e15 ), stored f16
__global__ __launch_bounds__(256) void k_att(const float* __restrict__ w1,
                                             const float* __restrict__ w2,
                                             const int* __restrict__ graph,
                                             half_t* __restrict__ att)
{
    __shared__ float red[256];
    const int row = blockIdx.x;           // (bl,i)
    const int bl = row >> 10;
    const int i  = row & (Nd - 1);
    const float e1 = w1[row];
    const int t = threadIdx.x;
    float ev[4];
    float mx = -3.4e38f;
    #pragma unroll
    for (int q = 0; q < 4; ++q) {
        int j = t + q * 256;
        float v = e1 + w2[(bl << 10) + j];
        v = (v >= 0.f) ? v : 0.2f * v;                 // leaky_relu 0.2
        if (graph[((size_t)i << 10) + j] <= 0) v = -9e15f;
        ev[q] = v;
        mx = fmaxf(mx, v);
    }
    red[t] = mx; __syncthreads();
    for (int st = 128; st > 0; st >>= 1) { if (t < st) red[t] = fmaxf(red[t], red[t+st]); __syncthreads(); }
    mx = red[0]; __syncthreads();
    float sum = 0.f;
    #pragma unroll
    for (int q = 0; q < 4; ++q) { ev[q] = __expf(ev[q] - mx); sum += ev[q]; }
    red[t] = sum; __syncthreads();
    for (int st = 128; st > 0; st >>= 1) { if (t < st) red[t] += red[t+st]; __syncthreads(); }
    const float inv = 1.f / red[0];
    half_t* dst = att + ((size_t)row << 10);
    #pragma unroll
    for (int q = 0; q < 4; ++q) dst[t + q * 256] = (half_t)(ev[q] * inv);
}

// fuse: x = sigmoid(alpha)*elu(hp1) + (1-s)*(gcn + b_mlp[c]) + x_in
__global__ void k_fuse(const float* __restrict__ hp1, const float* __restrict__ gcn,
                       const float* __restrict__ bmlp, const float* __restrict__ xres,
                       const float* __restrict__ alpha, float* __restrict__ out, size_t n)
{
    size_t i = (size_t)blockIdx.x * 256 + threadIdx.x;
    if (i >= n) return;
    int c = (int)(i & 255);
    float as = 1.f / (1.f + __expf(-alpha[0]));
    float g = hp1[i];
    g = (g > 0.f) ? g : (__expf(g) - 1.f);            // outer elu -> x_gat
    out[i] = as * g + (1.f - as) * (gcn[i] + bmlp[c]) + xres[i];
}

// LayerNorm over (n,c) per (b,l) slice (262144 elems); writes f32 + f16
__global__ __launch_bounds__(1024) void k_ln(const float* __restrict__ x,
                                             float* __restrict__ y, half_t* __restrict__ y16)
{
    __shared__ float rs[1024];
    __shared__ float rq[1024];
    const size_t base = (size_t)blockIdx.x * (Nd * Cd);
    const int t = threadIdx.x;
    float s = 0.f, q = 0.f;
    for (int i = t; i < Nd * Cd; i += 1024) { float v = x[base + i]; s += v; q += v * v; }
    rs[t] = s; rq[t] = q; __syncthreads();
    for (int st = 512; st > 0; st >>= 1) {
        if (t < st) { rs[t] += rs[t + st]; rq[t] += rq[t + st]; }
        __syncthreads();
    }
    const float inv_cnt = 1.f / (float)(Nd * Cd);
    const float mean = rs[0] * inv_cnt;
    float var = rq[0] * inv_cnt - mean * mean;
    var = fmaxf(var, 0.f);
    const float rstd = rsqrtf(var + 1e-5f);
    for (int i = t; i < Nd * Cd; i += 1024) {
        float v = (x[base + i] - mean) * rstd;
        y[base + i] = v;
        y16[base + i] = (half_t)v;
    }
}

// GLU gate: t = (g1+bg1)*sigmoid(g2+bg2) -> f16 (input to Wg3 GEMM)
__global__ void k_glu(const float* __restrict__ g1, const float* __restrict__ bg1,
                      const float* __restrict__ g2, const float* __restrict__ bg2,
                      half_t* __restrict__ t16, size_t n)
{
    size_t i = (size_t)blockIdx.x * 256 + threadIdx.x;
    if (i >= n) return;
    int c = (int)(i & 255);
    float a = g1[i] + bg1[c];
    float b = g2[i] + bg2[c];
    t16[i] = (half_t)(a * (1.f / (1.f + __expf(-b))));
}

// post: x2 = glu+bg3+y ; out = x2*weight[c,n] + bias[c,n] + x2
__global__ void k_post(const float* __restrict__ glu, const float* __restrict__ bg3,
                       const float* __restrict__ y, const float* __restrict__ w,
                       const float* __restrict__ bi, float* __restrict__ out, size_t n)
{
    size_t i = (size_t)blockIdx.x * 256 + threadIdx.x;
    if (i >= n) return;
    int c = (int)(i & 255);
    size_t r = i >> 8;
    int nn = (int)(r & (Nd - 1));
    float x2 = glu[i] + bg3[c] + y[i];
    out[i] = x2 * w[(size_t)c * Nd + nn] + bi[(size_t)c * Nd + nn] + x2;
}

// ---------------------------------------------------------------------------
extern "C" void kernel_launch(void* const* d_in, const int* in_sizes, int n_in,
                              void* d_out, int out_size, void* d_ws, size_t ws_size,
                              hipStream_t stream)
{
    const float* x_in  = (const float*)d_in[0];
    const int*   graph = (const int*)  d_in[1];
    const float* nv1   = (const float*)d_in[2];
    const float* nv2   = (const float*)d_in[3];
    const float* Wmlp  = (const float*)d_in[4];
    const float* bmlp  = (const float*)d_in[5];
    const float* Wg0   = (const float*)d_in[6];
    const float* ag0   = (const float*)d_in[7];
    const float* Wgo   = (const float*)d_in[8];
    const float* ago   = (const float*)d_in[9];
    const float* Wg1   = (const float*)d_in[10];
    const float* bg1   = (const float*)d_in[11];
    const float* Wg2   = (const float*)d_in[12];
    const float* bg2   = (const float*)d_in[13];
    const float* Wg3   = (const float*)d_in[14];
    const float* bg3   = (const float*)d_in[15];
    const float* alpha = (const float*)d_in[16];
    const float* wgt   = (const float*)d_in[17];
    const float* bia   = (const float*)d_in[18];

    const size_t RC  = (size_t)Rd * Cd;            // 12,582,912
    const size_t RC2 = (size_t)Rd * 512;

    // workspace layout
    size_t off = 0;
    auto walloc = [&](size_t bytes) { size_t o = off; off += (bytes + 255) & ~(size_t)255; return o; };
    char* ws = (char*)d_ws;
    size_t oHCAT  = walloc(RC2 * 2);               // f16 [R x 512] (A operand)
    size_t oXF32  = walloc(RC * 4);                // f32 [b,l,n,c] copy of x_in
    size_t oXT16  = walloc(RC * 2);                // f16 [bl][c][n]  (B of x1 GEMM)
    size_t oWHT   = walloc(RC * 2);                // f16 [bl][c][n]  (B of hp GEMMs)
    size_t oADPH  = walloc((size_t)Nd * Nd * 2);
    size_t oWG0T  = walloc((size_t)Cd * Cd * 2);   // Wg0^T
    size_t oWGOT  = walloc((size_t)Cd * Cd * 2);   // Wgo^T
    size_t oWMLPH = walloc((size_t)512 * Cd * 2);  // W_mlp as-is [o][c]
    size_t oWG1H  = walloc((size_t)Cd * Cd * 2);   // Wg1 as-is
    size_t oWG2H  = walloc((size_t)Cd * Cd * 2);
    size_t oWG3H  = walloc((size_t)Cd * Cd * 2);
    size_t oF32_1 = walloc(RC * 4);
    size_t oF32_2 = walloc(RC * 4);
    size_t oF32_3 = walloc(RC * 4);
    size_t oF32_4 = walloc(RC * 4);
    size_t oF16_1 = walloc(RC * 2);                // t16 (GLU gate, A operand)
    size_t oF16_2 = walloc(RC * 2);                // g0_16 / y16
    size_t oATT   = walloc((size_t)BLd * Nd * Nd * 2);
    size_t oV1    = walloc((size_t)Rd * 4);
    size_t oV2    = walloc((size_t)Rd * 4);
    if (off > ws_size) return;                     // not enough scratch

    half_t* HCAT  = (half_t*)(ws + oHCAT);
    float*  XF32  = (float*) (ws + oXF32);
    half_t* XT16  = (half_t*)(ws + oXT16);
    half_t* WHT   = (half_t*)(ws + oWHT);
    half_t* ADPH  = (half_t*)(ws + oADPH);
    half_t* WG0T  = (half_t*)(ws + oWG0T);
    half_t* WGOT  = (half_t*)(ws + oWGOT);
    half_t* WMLPH = (half_t*)(ws + oWMLPH);
    half_t* WG1H  = (half_t*)(ws + oWG1H);
    half_t* WG2H  = (half_t*)(ws + oWG2H);
    half_t* WG3H  = (half_t*)(ws + oWG3H);
    float*  F32_1 = (float*) (ws + oF32_1);
    float*  F32_2 = (float*) (ws + oF32_2);
    float*  F32_3 = (float*) (ws + oF32_3);
    float*  F32_4 = (float*) (ws + oF32_4);
    half_t* F16_1 = (half_t*)(ws + oF16_1);
    half_t* F16_2 = (half_t*)(ws + oF16_2);
    half_t* ATT   = (half_t*)(ws + oATT);
    float*  V1    = (float*) (ws + oV1);
    float*  V2    = (float*) (ws + oV2);

    auto gemm = [&](const half_t* A, const half_t* Bt, float* Cm,
                    int M, int Nn, int K, int lda, int ldbt, int ldc,
                    long long sA, long long sB, long long sC, int batches) {
        dim3 g(Nn / 64, M / 64, batches);
        gemm_wmma<<<g, dim3(256), 0, stream>>>(A, Bt, Cm, M, Nn, K, lda, ldbt, ldc, sA, sB, sC);
    };
    auto blocks = [](size_t n) { return (unsigned)((n + 255) / 256); };

    // --- weight prep: B operands stored column-major (BT[n][k]) ---
    k_cvtT<<<blocks(Cd*Cd),  256, 0, stream>>>(Wg0, WG0T, Cd, Cd, (size_t)Cd*Cd);   // [fo][fi]
    k_cvtT<<<blocks(Cd*Cd),  256, 0, stream>>>(Wgo, WGOT, Cd, Cd, (size_t)Cd*Cd);
    k_cvt <<<blocks(512*Cd), 256, 0, stream>>>(Wmlp, WMLPH, (size_t)512*Cd);        // [o][c]
    k_cvt <<<blocks(Cd*Cd),  256, 0, stream>>>(Wg1, WG1H, (size_t)Cd*Cd);
    k_cvt <<<blocks(Cd*Cd),  256, 0, stream>>>(Wg2, WG2H, (size_t)Cd*Cd);
    k_cvt <<<blocks(Cd*Cd),  256, 0, stream>>>(Wg3, WG3H, (size_t)Cd*Cd);

    // --- adaptive adjacency + input pack ---
    k_adp <<<Nd, 256, 0, stream>>>(nv1, nv2, ADPH);
    k_pack<<<blocks(RC), 256, 0, stream>>>(x_in, XF32, HCAT, XT16, RC);

    // --- GCN branch: x1 = adp @ X  (48 batches), then conv1x1 over 2C ---
    gemm(ADPH, XT16, F32_3, Nd, Cd, Nd, Nd, Nd, Cd,
         0LL, (long long)Cd * Nd, (long long)Nd * Cd, BLd);
    k_cvt_x1<<<blocks(RC), 256, 0, stream>>>(F32_3, HCAT, RC);
    gemm(HCAT, WMLPH, F32_3, Rd, Cd, 512, 512, 512, Cd, 0LL, 0LL, 0LL, 1);  // x_gcn

    // --- GAT layer 0 ---
    gemm(HCAT, WG0T, F32_1, Rd, Cd, Cd, 512, Cd, Cd, 0LL, 0LL, 0LL, 1);     // Wh0
    k_whT <<<blocks(RC), 256, 0, stream>>>(F32_1, WHT, RC);
    k_dots<<<Rd / 8, 256, 0, stream>>>(F32_1, ag0, V1, V2);
    k_att <<<Rd, 256, 0, stream>>>(V1, V2, graph, ATT);
    gemm(ATT, WHT, F32_2, Nd, Cd, Nd, Nd, Nd, Cd,
         (long long)Nd * Nd, (long long)Cd * Nd, (long long)Nd * Cd, BLd);  // hp0
    k_elu16<<<blocks(RC), 256, 0, stream>>>(F32_2, F16_2, RC);              // g0 (elu)

    // --- GAT layer 1 ---
    gemm(F16_2, WGOT, F32_1, Rd, Cd, Cd, Cd, Cd, Cd, 0LL, 0LL, 0LL, 1);     // Wh1
    k_whT <<<blocks(RC), 256, 0, stream>>>(F32_1, WHT, RC);
    k_dots<<<Rd / 8, 256, 0, stream>>>(F32_1, ago, V1, V2);
    k_att <<<Rd, 256, 0, stream>>>(V1, V2, graph, ATT);
    gemm(ATT, WHT, F32_2, Nd, Cd, Nd, Nd, Nd, Cd,
         (long long)Nd * Nd, (long long)Cd * Nd, (long long)Nd * Cd, BLd);  // hp1

    // --- fuse + layernorm ---
    k_fuse<<<blocks(RC), 256, 0, stream>>>(F32_2, F32_3, bmlp, XF32, alpha, F32_1, RC);
    k_ln  <<<BLd, 1024, 0, stream>>>(F32_1, F32_4, F16_2);                  // y, y16

    // --- GLU ---
    gemm(F16_2, WG1H, F32_3, Rd, Cd, Cd, Cd, Cd, Cd, 0LL, 0LL, 0LL, 1);     // g1
    gemm(F16_2, WG2H, F32_2, Rd, Cd, Cd, Cd, Cd, Cd, 0LL, 0LL, 0LL, 1);     // g2
    k_glu <<<blocks(RC), 256, 0, stream>>>(F32_3, bg1, F32_2, bg2, F16_1, RC);
    gemm(F16_1, WG3H, F32_1, Rd, Cd, Cd, Cd, Cd, Cd, 0LL, 0LL, 0LL, 1);     // glu_out

    // --- post affine + final layernorm + unpack ---
    k_post<<<blocks(RC), 256, 0, stream>>>(F32_1, bg3, F32_4, wgt, bia, F32_2, RC);
    k_ln  <<<BLd, 1024, 0, stream>>>(F32_2, F32_3, F16_2);
    k_unpack<<<blocks(RC), 256, 0, stream>>>(F32_3, (float*)d_out, RC);
}
